// M_LSTM_87024627351864
// MI455X (gfx1250) — matrix operand order repaired and checked by reading, not compiled, not run on hardware
//
#include <hip/hip_runtime.h>

// ---------------------------------------------------------------------------
// Shapes: V=32000, E=512, H=512, B=16, S=256, EH=1024, 4H=2048, N=B*S=4096
// All GEMMs are NT:  C[M,N] = A[M,K] * B[N,K]^T   (A and B rows K-contiguous)
// ---------------------------------------------------------------------------

typedef __attribute__((ext_vector_type(16))) __bf16 v16bf;
typedef __attribute__((ext_vector_type(8)))  __bf16 v8bf;
typedef __attribute__((ext_vector_type(8)))  float  v8f;

// ---- CDNA5 async global->LDS (ASYNCcnt) -----------------------------------
// ISA 10.x / 15.18: GLOBAL_LOAD_ASYNC_TO_LDS_B128, VDST = LDS byte-address
// VGPR, VADDR = 64-bit global address (GV mode, SADDR=off). Generic pointers
// to __shared__ carry the LDS byte offset in their low 32 bits (aperture
// mapping truncates addr[31:0]).
__device__ __forceinline__ unsigned lds_off32(const void* p) {
  return (unsigned)(unsigned long long)p;
}
__device__ __forceinline__ void async_copy_b128(unsigned lds_addr, const void* gptr) {
  asm volatile("global_load_async_to_lds_b128 %0, %1, off"
               :: "v"(lds_addr), "v"(gptr)
               : "memory");
}
__device__ __forceinline__ void wait_async_le3() {
#if __has_builtin(__builtin_amdgcn_s_wait_asynccnt)
  __builtin_amdgcn_s_wait_asynccnt(3);
#else
  asm volatile("s_wait_asynccnt 3" ::: "memory");
#endif
}
__device__ __forceinline__ void wait_async_0() {
#if __has_builtin(__builtin_amdgcn_s_wait_asynccnt)
  __builtin_amdgcn_s_wait_asynccnt(0);
#else
  asm volatile("s_wait_asynccnt 0" ::: "memory");
#endif
}

// ---- WMMA fragment loaders (wave32 layouts, CDNA5 ISA 7.12.2) -------------
// A 16x32 bf16: lanes 0-15 -> M=lane, K chunks [k0..k0+7] & [k0+16..k0+23]
//               lanes16-31 -> M=lane-16, K chunks [k0+8..15] & [k0+24..31]
__device__ __forceinline__ v16bf load_a_frag(const __bf16* base, int ld,
                                             int m0, int k0, int lane) {
  int row  = m0 + (lane & 15);
  int koff = k0 + ((lane >> 4) << 3);
  const __bf16* p = base + (size_t)row * ld + koff;
  v8bf lo = *(const v8bf*)p;
  v8bf hi = *(const v8bf*)(p + 16);
  return __builtin_shufflevector(lo, hi, 0,1,2,3,4,5,6,7,8,9,10,11,12,13,14,15);
}
// B 32x16 bf16 (from row-major W[N,K]): lane holds col N=lane&15,
// K chunks [k0+8h..+7] & [k0+16+8h..+7], h = lane>=16 (matches A).
__device__ __forceinline__ v16bf load_b_frag(const __bf16* base, int ld,
                                             int n0, int k0, int lane) {
  int row  = n0 + (lane & 15);
  int koff = k0 + ((lane >> 4) << 3);
  const __bf16* p = base + (size_t)row * ld + koff;
  v8bf lo = *(const v8bf*)p;
  v8bf hi = *(const v8bf*)(p + 16);
  return __builtin_shufflevector(lo, hi, 0,1,2,3,4,5,6,7,8,9,10,11,12,13,14,15);
}

// ---------------------------------------------------------------------------
// NT GEMM with double-buffered async global->LDS staging.
// 256 threads = 8 waves (4 m-waves x 2 n-waves), block tile 64(M) x 128(N),
// K step 32.  Per k-step each thread async-copies 16B of A and 2x16B of B
// into LDS (A tile 4KB, B tile 8KB); frags then come from ds_load_b128.
// ---------------------------------------------------------------------------
__global__ __launch_bounds__(256)
void gemm_nt_bf16(const __bf16* __restrict__ A, const __bf16* __restrict__ B,
                  const float* __restrict__ bias,
                  float* __restrict__ Cf, __bf16* __restrict__ Cb,
                  int M, int N, int K) {
  __shared__ __bf16 sA[2][64 * 32];    // 2 x 4KB
  __shared__ __bf16 sB[2][128 * 32];   // 2 x 8KB

  const int tid  = threadIdx.x;
  const int lane = tid & 31;
  const int wave = tid >> 5;
  const int wm = wave & 3, wn = wave >> 2;
  const int m0b = blockIdx.x * 64;
  const int n0b = blockIdx.y * 128;

  // staging coords: 16B (8 bf16) chunk per thread; A: 64 rows x 4 chunks,
  // B: 128 rows x 4 chunks (two rounds).
  const int srow = tid >> 2;
  const int scol = (tid & 3) << 3;

  const __bf16* gA  = A + (size_t)(m0b + srow) * K + scol;
  const __bf16* gB0 = B + (size_t)(n0b + srow) * K + scol;
  const __bf16* gB1 = B + (size_t)(n0b + 64 + srow) * K + scol;

  const unsigned lA  = lds_off32(&sA[0][srow * 32 + scol]);
  const unsigned lB0 = lds_off32(&sB[0][srow * 32 + scol]);
  const unsigned lB1 = lds_off32(&sB[0][(64 + srow) * 32 + scol]);
  const unsigned bufAstride = (unsigned)(64 * 32 * sizeof(__bf16));
  const unsigned bufBstride = (unsigned)(128 * 32 * sizeof(__bf16));

  v8f acc[4];
  acc[0] = (v8f)0.f; acc[1] = (v8f)0.f; acc[2] = (v8f)0.f; acc[3] = (v8f)0.f;

  const int nk = K >> 5;

  // prologue: stage k-chunk 0 into buffer 0
  async_copy_b128(lA,  gA);
  async_copy_b128(lB0, gB0);
  async_copy_b128(lB1, gB1);

  for (int kt = 0; kt < nk; ++kt) {
    const int cur = kt & 1;
    if (kt + 1 < nk) {
      const int k1 = (kt + 1) << 5;
      const int nxt = cur ^ 1;
      // deep prefetch to L2 for the chunk after next
      __builtin_prefetch((const void*)(gB0 + k1 + 32), 0, 1);
      async_copy_b128(lA  + nxt * bufAstride, gA  + k1);
      async_copy_b128(lB0 + nxt * bufBstride, gB0 + k1);
      async_copy_b128(lB1 + nxt * bufBstride, gB1 + k1);
      wait_async_le3();   // the 3 newest (next buffer) may stay in flight
    } else {
      wait_async_0();
    }
    __syncthreads();      // publish LDS tile across waves

    v16bf a = load_a_frag(&sA[cur][0], 32, wm * 16, 0, lane);
#pragma unroll
    for (int t = 0; t < 4; ++t) {
      v16bf b = load_b_frag(&sB[cur][0], 32, wn * 64 + t * 16, 0, lane);
      acc[t] = __builtin_amdgcn_wmma_f32_16x16x32_bf16(
          false, a, false, b, (short)0, acc[t], false, false);
    }
    __syncthreads();      // tile consumed; safe to overwrite next iteration
  }

  const int hi = lane >> 4, ln = lane & 15;
#pragma unroll
  for (int t = 0; t < 4; ++t) {
    int col = n0b + wn * 64 + t * 16 + ln;
    float bv = bias ? bias[col] : 0.f;
#pragma unroll
    for (int r = 0; r < 8; ++r) {
      int row = m0b + wm * 16 + r + hi * 8;
      float v = acc[t][r] + bv;
      if (Cf) Cf[(size_t)row * N + col] = v;
      else    Cb[(size_t)row * N + col] = (__bf16)v;
    }
  }
}

// ---------------------------------------------------------------------------
// Embedding gather -> bf16, stored [t*16+b][512] (time-major for the scan).
// ---------------------------------------------------------------------------
__global__ void embed_gather(const int* __restrict__ ids,
                             const float* __restrict__ emb,
                             __bf16* __restrict__ x) {
  int n = blockIdx.x;           // n = t*16 + b
  int b = n & 15, t = n >> 4;
  int id = ids[b * 256 + t];
  const float* row = emb + (size_t)id * 512;
  for (int e = threadIdx.x; e < 512; e += blockDim.x)
    x[(size_t)n * 512 + e] = (__bf16)row[e];
}

// Split concat([W_i,W_f,W_o,W_a]) [2048,1024] into Wx[2048,512], Wh[2048,512] bf16.
__global__ void w4_split(const float* __restrict__ Wi, const float* __restrict__ Wf,
                         const float* __restrict__ Wo, const float* __restrict__ Wa,
                         __bf16* __restrict__ Wx, __bf16* __restrict__ Wh) {
  int j = blockIdx.x;           // 0..2047
  const float* src[4] = {Wi, Wf, Wo, Wa};
  const float* W = src[j >> 9];
  int r = j & 511;
  for (int e = threadIdx.x; e < 512; e += blockDim.x) {
    Wx[(size_t)j * 512 + e] = (__bf16)W[r * 1024 + e];
    Wh[(size_t)j * 512 + e] = (__bf16)W[r * 1024 + 512 + e];
  }
}

__global__ void f32_to_bf16(const float* __restrict__ src, __bf16* __restrict__ dst, int n) {
  int stride = gridDim.x * blockDim.x;
  for (int i = blockIdx.x * blockDim.x + threadIdx.x; i < n; i += stride)
    dst[i] = (__bf16)src[i];
}

// ---------------------------------------------------------------------------
// LSTM scan: ONE persistent workgroup, 1024 threads = 32 waves.
// Wave w owns gate-output columns [w*64, w*64+64) of the 2048 (gate g=w/8).
// Per step: z = zx[t] + h * Wh^T (WMMA), LayerNorm per (b,gate), sigmoid/gelu,
// cell update; h kept in LDS as bf16 (A-side of next step's WMMA).
// Dynamic LDS: h_s 16KB | c_s 32KB | act 128KB | red 1KB  (=181248 B < 320KB)
// ---------------------------------------------------------------------------
__global__ __launch_bounds__(1024)
void lstm_scan(const __bf16* __restrict__ Wh, const float* __restrict__ zx,
               const float* __restrict__ h0,
               const float* __restrict__ g_i, const float* __restrict__ b_i,
               const float* __restrict__ g_f, const float* __restrict__ b_f,
               const float* __restrict__ g_o, const float* __restrict__ b_o,
               const float* __restrict__ g_a, const float* __restrict__ b_a,
               __bf16* __restrict__ hs) {
  extern __shared__ char smem[];
  __bf16* h_s = (__bf16*)smem;                          // [16][512]
  float*  c_s = (float*)(smem + 16384);                 // [16][512]
  float*  act = (float*)(smem + 16384 + 32768);         // [4][16][512]
  float*  red = (float*)(smem + 16384 + 32768 + 131072);// sum 64 | sq 64 | mu 64 | rs 64

  const int tid  = threadIdx.x;
  const int lane = tid & 31;
  const int w    = tid >> 5;          // wave 0..31
  const int hi   = lane >> 4, ln = lane & 15;
  const int g    = w >> 3;            // gate of this wave
  const float* gp[4] = {g_i, g_f, g_o, g_a};
  const float* bp[4] = {b_i, b_f, b_o, b_a};
  const float* gw = gp[g];
  const float* bw = bp[g];

  // init h = broadcast(h0), c = h  (reference: c_init = h_init)
  for (int idx = tid; idx < 16 * 512; idx += 1024) {
    float v = h0[idx & 511];
    h_s[idx] = (__bf16)v;
    c_s[idx] = v;
  }
  __syncthreads();

  for (int t = 0; t < 256; ++t) {
    if (tid < 128) red[tid] = 0.f;    // clear sum/sumsq
    __syncthreads();

    // ---- z = zx[t] + h * Wh^T : wave computes 16 x 64 (4 tiles), K=512 ----
    v8f acc[4];
#pragma unroll
    for (int tt = 0; tt < 4; ++tt) {
      int col = w * 64 + tt * 16 + ln;
#pragma unroll
      for (int r = 0; r < 8; ++r)
        acc[tt][r] = zx[(size_t)(t * 16 + r + hi * 8) * 2048 + col];
    }
#pragma unroll
    for (int k0 = 0; k0 < 512; k0 += 32) {
      v16bf a = load_a_frag(h_s, 512, 0, k0, lane);   // from LDS (ds_load_b128)
#pragma unroll
      for (int tt = 0; tt < 4; ++tt) {
        v16bf b = load_b_frag(Wh, 512, w * 64 + tt * 16, k0, lane);
        acc[tt] = __builtin_amdgcn_wmma_f32_16x16x32_bf16(
            false, a, false, b, (short)0, acc[tt], false, false);
      }
    }

    // ---- LayerNorm stats: per-(b,gate) sums over this wave's 64 cols ------
#pragma unroll
    for (int r = 0; r < 8; ++r) {
      float s = acc[0][r] + acc[1][r] + acc[2][r] + acc[3][r];
      float q = acc[0][r]*acc[0][r] + acc[1][r]*acc[1][r]
              + acc[2][r]*acc[2][r] + acc[3][r]*acc[3][r];
#pragma unroll
      for (int off = 8; off >= 1; off >>= 1) {
        s += __shfl_xor(s, off, 16);
        q += __shfl_xor(q, off, 16);
      }
      if (ln == 0) {
        int m = r + hi * 8;
        atomicAdd(&red[g * 16 + m], s);        // ds_add_f32
        atomicAdd(&red[64 + g * 16 + m], q);
      }
    }
    __syncthreads();

    if (tid < 64) {                    // mean / rstd per (gate, b)
      float mu  = red[tid] * (1.f / 512.f);
      float var = red[64 + tid] * (1.f / 512.f) - mu * mu;
      red[128 + tid] = mu;
      red[192 + tid] = rsqrtf(var + 1e-5f);
    }
    __syncthreads();

    // ---- LN + activation -> act[g][b][col] --------------------------------
#pragma unroll
    for (int tt = 0; tt < 4; ++tt) {
      int j  = w * 64 + tt * 16 + ln;
      int cg = j & 511;
      float gg = gw[cg], bb = bw[cg];
#pragma unroll
      for (int r = 0; r < 8; ++r) {
        int m = r + hi * 8;
        float mu = red[128 + g * 16 + m];
        float rs = red[192 + g * 16 + m];
        float zn = (acc[tt][r] - mu) * rs * gg + bb;
        float av = (g == 3) ? 0.5f * zn * (1.f + erff(zn * 0.70710678f))
                            : 1.f / (1.f + expf(-zn));
        act[(g * 16 + m) * 512 + cg] = av;
      }
    }
    __syncthreads();

    // ---- cell update: c2 = c*f + i*a ; h2 = c2*o --------------------------
    for (int idx = tid; idx < 16 * 512; idx += 1024) {
      int b = idx >> 9, col = idx & 511;
      float iv = act[b * 512 + col];
      float fv = act[8192  + b * 512 + col];
      float ov = act[16384 + b * 512 + col];
      float av = act[24576 + b * 512 + col];
      float c2 = c_s[idx] * fv + iv * av;
      float h2 = c2 * ov;
      c_s[idx] = c2;
      h_s[idx] = (__bf16)h2;
      hs[((size_t)b * 256 + t) * 512 + col] = (__bf16)h2;   // [b*S+t] order
    }
    __syncthreads();
  }
}

// ---------------------------------------------------------------------------
extern "C" void kernel_launch(void* const* d_in, const int* in_sizes, int n_in,
                              void* d_out, int out_size, void* d_ws, size_t ws_size,
                              hipStream_t stream) {
  const int*   ids  = (const int*)  d_in[0];
  const float* emb  = (const float*)d_in[1];
  const float* W_i  = (const float*)d_in[2];
  const float* g_i  = (const float*)d_in[3];
  const float* b_i  = (const float*)d_in[4];
  const float* W_f  = (const float*)d_in[5];
  const float* g_f  = (const float*)d_in[6];
  const float* b_f  = (const float*)d_in[7];
  const float* W_o  = (const float*)d_in[8];
  const float* g_o  = (const float*)d_in[9];
  const float* b_o  = (const float*)d_in[10];
  const float* W_a  = (const float*)d_in[11];
  const float* g_a  = (const float*)d_in[12];
  const float* b_a  = (const float*)d_in[13];
  const float* Wc1  = (const float*)d_in[14];
  const float* bout = (const float*)d_in[15];
  const float* h0   = (const float*)d_in[16];
  float* logits = (float*)d_out;

  // workspace carve (all offsets 512B-aligned)
  char* w = (char*)d_ws;
  __bf16* x_bf   = (__bf16*)(w);                 // [4096,512]   4 MB  (time-major)
  __bf16* Wx_bf  = (__bf16*)(w + 4194304);       // [2048,512]   2 MB
  __bf16* Wh_bf  = (__bf16*)(w + 6291456);       // [2048,512]   2 MB
  __bf16* emb_bf = (__bf16*)(w + 8388608);       // [32000,512] 32.77 MB
  __bf16* wc1_bf = (__bf16*)(w + 41156608);      // [512,512]   0.5 MB
  float*  zx     = (float*) (w + 41680896);      // [256,16,2048] 33.55 MB
  __bf16* hs_bf  = (__bf16*)(w + 75235328);      // [4096,512]   4 MB  (b*S+t order)
  __bf16* pr_bf  = (__bf16*)(w + 79429632);      // [4096,512]   4 MB

  // 1) prep: gather + bf16 conversions
  embed_gather<<<4096, 256, 0, stream>>>(ids, emb, x_bf);
  w4_split<<<2048, 256, 0, stream>>>(W_i, W_f, W_o, W_a, Wx_bf, Wh_bf);
  f32_to_bf16<<<4096, 256, 0, stream>>>(emb, emb_bf, 32000 * 512);
  f32_to_bf16<<<256, 256, 0, stream>>>(Wc1, wc1_bf, 512 * 512);

  // 2) hoisted input-gate GEMM: zx[4096,2048] = x * Wx^T
  gemm_nt_bf16<<<dim3(64, 16), 256, 0, stream>>>(
      x_bf, Wx_bf, nullptr, zx, nullptr, 4096, 2048, 512);

  // 3) serial recurrence (single persistent WGP, 181 KB dynamic LDS)
  lstm_scan<<<1, 1024, 181248, stream>>>(
      Wh_bf, zx, h0, g_i, b_i, g_f, b_f, g_o, b_o, g_a, b_a, hs_bf);

  // 4) proj[4096,512] = hs * Wc1^T  (bf16 out for next GEMM)
  gemm_nt_bf16<<<dim3(64, 4), 256, 0, stream>>>(
      hs_bf, wc1_bf, nullptr, nullptr, pr_bf, 4096, 512, 512);

  // 5) logits[4096,32000] = proj * emb^T + b_out
  gemm_nt_bf16<<<dim3(64, 250), 256, 0, stream>>>(
      pr_bf, emb_bf, bout, logits, nullptr, 4096, 32000, 512);
}
